// Q2A_Interaction_Global_90915867722508
// MI455X (gfx1250) — matrix operand order repaired and checked by reading, not compiled, not run on hardware
//
#include <hip/hip_runtime.h>
#include <cstdint>
#include <cstddef>

// ---------------------------------------------------------------------------
// Problem constants (from reference)
// ---------------------------------------------------------------------------
constexpr int cB = 8, cQ = 8, cA = 32, cS = 16, cT = 4096;
constexpr int cTD = 768, cVITD = 1024, cDS = 512, cNB = 3, cPRE = 3584;

// ---------------------------------------------------------------------------
// WMMA types (CDNA5 gfx1250, wave32)
// ---------------------------------------------------------------------------
typedef __attribute__((ext_vector_type(16))) _Float16 v16h;
typedef __attribute__((ext_vector_type(8)))  float    v8f;

union FragH { uint4 u[2]; v16h v; };      // 16 x f16 = 32B = 2 x uint4
union Pack4 { _Float16 h[4]; uint2 u; };  // 4 x f16  = 8B

// ---------------------------------------------------------------------------
// Generic tiled GEMM:  C[M,N] = act(A[M,K] @ W[K,N] + bias[N])
// fp32 in / fp32 out, f16 WMMA compute (v_wmma_f32_16x16x32_f16).
// Tile 128x128x32, 256 threads = 8 waves (2x4 wave grid), each wave 64Mx32N.
// LDS double-buffered: global loads for step s+1 are issued before the 8
// WMMAs of step s; converted tiles land in the alternate buffer, one
// barrier per K-step.  A staged [m][k] f16, W staged transposed [n][k] f16
// so each 16x16x32 fragment is two ds_load_b128 per lane, matching the ISA
// 16-bit A/B VGPR layouts (lanes 0-15 / 16-31 K-split).
// Requires: N % 128 == 0, K % 32 == 0 (true for every call below). M guarded.
// ---------------------------------------------------------------------------
#define BM 128
#define BN 128
#define BK 32

template <int RELU>
__global__ __launch_bounds__(256)
void k_gemm(const float* __restrict__ Ag, const float* __restrict__ Wg,
            const float* __restrict__ bias, float* __restrict__ Cg,
            int M, int N, int K)
{
    __shared__ _Float16 As[2][BM * BK];   // [m][k]
    __shared__ _Float16 Bs[2][BN * BK];   // [n][k]  (transposed W tile)

    const int tid  = threadIdx.x;
    const int lane = tid & 31;
    const int wave = tid >> 5;
    const int wm   = wave >> 2;        // 0..1 : M group of 4 tiles
    const int wn   = wave & 3;         // 0..3 : N group of 2 tiles
    const int half = lane >> 4;        // half-wave (K split per ISA layout)
    const int l16  = lane & 15;
    const int m0   = blockIdx.x * BM;
    const int n0   = blockIdx.y * BN;
    const bool full = (m0 + BM <= M);  // block-uniform: interior fast path

    // per-thread staging coordinates
    const int arow[4] = { (tid + 0*256) >> 3, (tid + 1*256) >> 3,
                          (tid + 2*256) >> 3, (tid + 3*256) >> 3 };
    const int acol    = (tid & 7) * 4;
    const int brow[4] = { (tid + 0*256) >> 5, (tid + 1*256) >> 5,
                          (tid + 2*256) >> 5, (tid + 3*256) >> 5 };
    const int bcol    = (tid & 31) * 4;

    float4 ra[4], rb[4];               // register-staged tiles (in flight)

    auto gload = [&](int kk) {
        if (full) {
#pragma unroll
            for (int i = 0; i < 4; ++i)
                ra[i] = *(const float4*)(Ag + (size_t)(m0 + arow[i]) * K + kk + acol);
        } else {
#pragma unroll
            for (int i = 0; i < 4; ++i) {
                ra[i] = (float4){0.f, 0.f, 0.f, 0.f};
                if (m0 + arow[i] < M)
                    ra[i] = *(const float4*)(Ag + (size_t)(m0 + arow[i]) * K + kk + acol);
            }
        }
#pragma unroll
        for (int i = 0; i < 4; ++i)
            rb[i] = *(const float4*)(Wg + (size_t)(kk + brow[i]) * N + n0 + bcol);
        // keep L2 warm one more tile ahead
        if (kk + BK < K)
            __builtin_prefetch(Wg + (size_t)(kk + BK) * N + n0 + lane * 4, 0, 1);
    };

    auto lstore = [&](int buf) {
#pragma unroll
        for (int i = 0; i < 4; ++i) {
            Pack4 p;
            p.h[0] = (_Float16)ra[i].x; p.h[1] = (_Float16)ra[i].y;
            p.h[2] = (_Float16)ra[i].z; p.h[3] = (_Float16)ra[i].w;
            *(uint2*)(As[buf] + arow[i] * BK + acol) = p.u;
        }
#pragma unroll
        for (int i = 0; i < 4; ++i) {
            Bs[buf][(bcol + 0) * BK + brow[i]] = (_Float16)rb[i].x;
            Bs[buf][(bcol + 1) * BK + brow[i]] = (_Float16)rb[i].y;
            Bs[buf][(bcol + 2) * BK + brow[i]] = (_Float16)rb[i].z;
            Bs[buf][(bcol + 3) * BK + brow[i]] = (_Float16)rb[i].w;
        }
    };

    v8f acc[4][2];
#pragma unroll
    for (int i = 0; i < 4; ++i)
#pragma unroll
        for (int j = 0; j < 2; ++j)
            acc[i][j] = (v8f){0.f, 0.f, 0.f, 0.f, 0.f, 0.f, 0.f, 0.f};

    const int nsteps = K / BK;
    gload(0);
    lstore(0);
    __syncthreads();

    for (int s = 0; s < nsteps; ++s) {
        const int cur = s & 1;
        if (s + 1 < nsteps) gload((s + 1) * BK);   // overlap with WMMAs below

        // ---- load fragments (two b128 LDS loads each, per ISA layout) ----
        FragH fa[4], fb[2];
#pragma unroll
        for (int i = 0; i < 4; ++i) {
            const _Float16* src = As[cur] + ((wm * 4 + i) * 16 + l16) * BK;
            fa[i].u[0] = *(const uint4*)(src + half * 8);        // K 0-7 / 8-15
            fa[i].u[1] = *(const uint4*)(src + 16 + half * 8);   // K 16-23 / 24-31
        }
#pragma unroll
        for (int j = 0; j < 2; ++j) {
            const _Float16* src = Bs[cur] + ((wn * 2 + j) * 16 + l16) * BK;
            fb[j].u[0] = *(const uint4*)(src + half * 8);        // K 0-7 / 8-15
            fb[j].u[1] = *(const uint4*)(src + 16 + half * 8);   // K 16-23 / 24-31
        }
        // ---- 8 x v_wmma_f32_16x16x32_f16 ----
#pragma unroll
        for (int i = 0; i < 4; ++i)
#pragma unroll
            for (int j = 0; j < 2; ++j)
                acc[i][j] = __builtin_amdgcn_wmma_f32_16x16x32_f16(
                    false, fa[i].v, false, fb[j].v,
                    (short)0, acc[i][j], false, false);

        if (s + 1 < nsteps) {
            lstore((s + 1) & 1);   // write alternate buffer (no reader conflict)
            __syncthreads();
        }
    }

    // ---- epilogue: C/D layout -> lanes 0-15 M=r, lanes 16-31 M=8+r ----
#pragma unroll
    for (int j = 0; j < 2; ++j) {
        int col  = n0 + (wn * 2 + j) * 16 + l16;
        float bv = bias ? bias[col] : 0.f;
#pragma unroll
        for (int i = 0; i < 4; ++i) {
#pragma unroll
            for (int r = 0; r < 8; ++r) {
                int row = m0 + (wm * 4 + i) * 16 + half * 8 + r;
                float v = acc[i][j][r] + bv;
                if (RELU) v = fmaxf(v, 0.f);
                if (full || row < M)
                    Cg[(size_t)row * N + col] = v;
            }
        }
    }
}

// ---------------------------------------------------------------------------
// softmax over S=16 para scores (one thread per batch)
// ---------------------------------------------------------------------------
__global__ void k_softmax16(const float* __restrict__ ps, float* __restrict__ score)
{
    int b = threadIdx.x;
    if (b < cB) {
        float v[cS], m = -1e30f;
        for (int s = 0; s < cS; ++s) { v[s] = ps[b * cS + s]; m = fmaxf(m, v[s]); }
        float z = 0.f;
        for (int s = 0; s < cS; ++s) { v[s] = __expf(v[s] - m); z += v[s]; }
        float inv = 1.f / z;
        for (int s = 0; s < cS; ++s) score[b * cS + s] = v[s] * inv;
    }
}

// ---------------------------------------------------------------------------
// out[b,d] = sum_s score[b,s] * X[b,s,d]    grid (B, D/256)
// ---------------------------------------------------------------------------
__global__ void k_wsum(const float* __restrict__ score, const float* __restrict__ X,
                       float* __restrict__ out)
{
    int b = blockIdx.x;
    int d = blockIdx.y * 256 + threadIdx.x;
    float acc = 0.f;
    for (int s = 0; s < cS; ++s)
        acc += score[b * cS + s] * X[((size_t)b * cS + s) * cDS + d];
    out[(size_t)b * cDS + d] = acc;
}

// ---------------------------------------------------------------------------
// Per-segment softmax(hand_score) pooling over frames.
// Empty segment (end<=start) is exactly the range [start, start+1).
// One block (512 threads = 16 waves) per (b,s); thread d owns feature d.
// ---------------------------------------------------------------------------
__global__ __launch_bounds__(512)
void k_seg_pool(const float* __restrict__ hand, const int* __restrict__ segs,
                const int* __restrict__ sege, const float* __restrict__ vid,
                float* __restrict__ seg_feat)
{
    __shared__ float red[16];
    __shared__ float sM, sZ;
    int bs = blockIdx.x, b = bs >> 4;     // S = 16
    int tid = threadIdx.x, lane = tid & 31, wv = tid >> 5;
    int lo = segs[bs], e = sege[bs];
    int hi = (e <= lo) ? (lo + 1) : e;
    const float* h = hand + (size_t)b * cT;

    // max
    float m = -1e30f;
    for (int t = lo + tid; t < hi; t += 512) m = fmaxf(m, h[t]);
#pragma unroll
    for (int o = 16; o >= 1; o >>= 1) m = fmaxf(m, __shfl_xor(m, o, 32));
    if (lane == 0) red[wv] = m;
    __syncthreads();
    if (tid == 0) {
        float mm = red[0];
        for (int i = 1; i < 16; ++i) mm = fmaxf(mm, red[i]);
        sM = mm;
    }
    __syncthreads();
    float mM = sM;

    // sum exp
    float z = 0.f;
    for (int t = lo + tid; t < hi; t += 512) z += __expf(h[t] - mM);
#pragma unroll
    for (int o = 16; o >= 1; o >>= 1) z += __shfl_xor(z, o, 32);
    if (lane == 0) red[wv] = z;
    __syncthreads();
    if (tid == 0) {
        float zz = 0.f;
        for (int i = 0; i < 16; ++i) zz += red[i];
        sZ = zz;
    }
    __syncthreads();
    float inv = 1.f / sZ;

    // weighted feature accumulation: thread tid owns feature d = tid (DS=512)
    int d = tid;
    float acc = 0.f;
    for (int t = lo; t < hi; ++t)
        acc += __expf(h[t] - mM) * vid[((size_t)b * cT + t) * cDS + d];
    seg_feat[(size_t)bs * cDS + d] = acc * inv;
}

// qa = at + broadcast(q)
__global__ void k_addq(const float* __restrict__ at, const float* __restrict__ qv,
                       float* __restrict__ qa)
{
    size_t i = (size_t)blockIdx.x * 256 + threadIdx.x;
    const size_t tot = (size_t)cB * cQ * cA * cDS;
    if (i < tot) {
        int b = (int)(i / ((size_t)cQ * cA * cDS));
        int d = (int)(i & (cDS - 1));
        qa[i] = at[i] + qv[b * cDS + d];
    }
}

// pre_in[row, :] = [video_seg(b) | vsg(b) | text_seg(b) | qa(row) | ab(row, 3*DS)]
__global__ void k_concat(const float* __restrict__ video_seg, const float* __restrict__ vsg,
                         const float* __restrict__ text_seg, const float* __restrict__ qa,
                         const float* __restrict__ ab, float* __restrict__ pre_in)
{
    size_t i = (size_t)blockIdx.x * 256 + threadIdx.x;
    const size_t tot = (size_t)cB * cQ * cA * cPRE;
    if (i >= tot) return;
    int row = (int)(i / cPRE);
    int col = (int)(i % cPRE);
    int b   = row / (cQ * cA);
    float v;
    if      (col <     cDS) v = video_seg[b * cDS + col];
    else if (col < 2 * cDS) v = vsg[b * cDS + col - cDS];
    else if (col < 3 * cDS) v = text_seg[b * cDS + col - 2 * cDS];
    else if (col < 4 * cDS) v = qa[(size_t)row * cDS + col - 3 * cDS];
    else                    v = ab[(size_t)row * (cNB * cDS) + (col - 4 * cDS)];
    pre_in[i] = v;
}

// ---------------------------------------------------------------------------
// Sequential GRU + proj + argmax state feedback. One block per batch.
// State / gh / hnew kept in LDS across the Q=8 steps (hnew as f16, ~40KB LDS).
// ---------------------------------------------------------------------------
__global__ __launch_bounds__(256)
void k_gru(const float* __restrict__ pre, const float* __restrict__ Wih,
           const float* __restrict__ Whh, const float* __restrict__ bih,
           const float* __restrict__ bhh, const float* __restrict__ W1p,
           const float* __restrict__ b1p, const float* __restrict__ w2p,
           const float* __restrict__ b2p, const float* __restrict__ state0,
           float* __restrict__ out)
{
    __shared__ float     s_state[cDS];
    __shared__ float     s_gh[3 * cDS];
    __shared__ _Float16  s_hnew[cA * cDS];
    __shared__ float     s_logits[cA];
    __shared__ int       s_am;

    int b = blockIdx.x, tid = threadIdx.x;
    for (int d = tid; d < cDS; d += 256) s_state[d] = state0[d];
    __syncthreads();

    for (int qs = 0; qs < cQ; ++qs) {
        const float* x = pre + ((size_t)(b * cQ + qs)) * cA * cDS;

        // gh = state @ Whh + bhh   (state identical across rows)
        for (int j = tid; j < 3 * cDS; j += 256) {
            float acc = bhh[j];
            for (int k = 0; k < cDS; ++k)
                acc += s_state[k] * Whh[(size_t)k * (3 * cDS) + j];
            s_gh[j] = acc;
        }
        if (tid < cA) s_logits[tid] = 0.f;
        __syncthreads();

        // gates + hnew
        for (int idx = tid; idx < cA * cDS; idx += 256) {
            int a = idx >> 9, d = idx & 511;
            const float* xa = x + (size_t)a * cDS;
            float ir = bih[d], iz = bih[d + cDS], inn = bih[d + 2 * cDS];
            for (int k = 0; k < cDS; ++k) {
                float xv = xa[k];
                const float* wr = Wih + (size_t)k * (3 * cDS);
                ir  += xv * wr[d];
                iz  += xv * wr[d + cDS];
                inn += xv * wr[d + 2 * cDS];
            }
            float r  = 1.f / (1.f + __expf(-(ir + s_gh[d])));
            float zt = 1.f / (1.f + __expf(-(iz + s_gh[d + cDS])));
            float n  = tanhf(inn + r * s_gh[d + 2 * cDS]);
            float hv = (1.f - zt) * n + zt * s_state[d];
            s_hnew[idx] = (_Float16)hv;
        }
        __syncthreads();

        // logits = relu(hnew @ W1p + b1p) @ w2p   (block-cooperative per row)
        for (int a = 0; a < cA; ++a) {
            const _Float16* hv = s_hnew + a * cDS;
            float part = 0.f;
            for (int n = tid; n < 2048; n += 256) {
                float hs = b1p[n];
                for (int d = 0; d < cDS; ++d)
                    hs += (float)hv[d] * W1p[(size_t)d * 2048 + n];
                part += w2p[n] * fmaxf(hs, 0.f);
            }
            atomicAdd(&s_logits[a], part);
        }
        __syncthreads();

        if (tid == 0) {
            int am = 0; float bm = s_logits[0];
            for (int a = 1; a < cA; ++a)
                if (s_logits[a] > bm) { bm = s_logits[a]; am = a; }
            s_am = am;
        }
        __syncthreads();
        if (tid < cA)
            out[((size_t)(b * cQ + qs)) * cA + tid] = s_logits[tid] + b2p[0];
        int am = s_am;
        for (int d = tid; d < cDS; d += 256)
            s_state[d] = x[(size_t)am * cDS + d];
        __syncthreads();
    }
}

// ---------------------------------------------------------------------------
// Host driver
// ---------------------------------------------------------------------------
extern "C" void kernel_launch(void* const* d_in, const int* in_sizes, int n_in,
                              void* d_out, int out_size, void* d_ws, size_t ws_size,
                              hipStream_t stream)
{
    (void)in_sizes; (void)n_in; (void)out_size; (void)ws_size;

    // setup_inputs order: video, video_global, para, question, a_texts,
    // a_buttons, paras_score, hand_score, params{mlp_t(4), mlp_v(4),
    // mlp_vit(4), mlp_pre(4), gru(4), proj(4), state}, seg_start, seg_end
    const float* video      = (const float*)d_in[0];
    const float* video_glob = (const float*)d_in[1];
    const float* para       = (const float*)d_in[2];
    const float* question   = (const float*)d_in[3];
    const float* a_texts    = (const float*)d_in[4];
    const float* a_buttons  = (const float*)d_in[5];
    const float* paras_sc   = (const float*)d_in[6];
    const float* hand_sc    = (const float*)d_in[7];
    const float* Wt1   = (const float*)d_in[8];
    const float* bt1   = (const float*)d_in[9];
    const float* Wt2   = (const float*)d_in[10];
    const float* bt2   = (const float*)d_in[11];
    const float* Wv1   = (const float*)d_in[12];
    const float* bv1   = (const float*)d_in[13];
    const float* Wv2   = (const float*)d_in[14];
    const float* bv2   = (const float*)d_in[15];
    const float* Wvit1 = (const float*)d_in[16];
    const float* bvit1 = (const float*)d_in[17];
    const float* Wvit2 = (const float*)d_in[18];
    const float* bvit2 = (const float*)d_in[19];
    const float* Wp1   = (const float*)d_in[20];
    const float* bp1   = (const float*)d_in[21];
    const float* Wp2   = (const float*)d_in[22];
    const float* bp2   = (const float*)d_in[23];
    const float* Wih   = (const float*)d_in[24];
    const float* Whh   = (const float*)d_in[25];
    const float* bihp  = (const float*)d_in[26];
    const float* bhhp  = (const float*)d_in[27];
    const float* Wpr1  = (const float*)d_in[28];
    const float* bpr1  = (const float*)d_in[29];
    const float* wpr2  = (const float*)d_in[30];
    const float* bpr2  = (const float*)d_in[31];
    const float* state0 = (const float*)d_in[32];
    const int*   seg_start = (const int*)d_in[33];
    const int*   seg_end   = (const int*)d_in[34];
    float* out = (float*)d_out;

    // --- workspace carve (~230 MB, with region reuse) ---
    char* base = (char*)d_ws;
    size_t off = 0;
    auto take = [&](size_t nfloat) -> float* {
        float* p = (float*)(base + off);
        off += (nfloat * sizeof(float) + 255) & ~(size_t)255;
        return p;
    };
    float* score     = take((size_t)cB * cS);
    float* text_seg  = take((size_t)cB * cDS);
    float* vsg       = take((size_t)cB * cDS);
    float* video_seg = take((size_t)cB * cDS);
    float* qv        = take((size_t)cB * cDS);
    float* seg_feat  = take((size_t)cB * cS * cDS);
    float* para_f    = take((size_t)cB * cS * cDS);
    float* vg        = take((size_t)cB * cS * cDS);
    float* hbuf      = take((size_t)128 * 3072);          // para_h / vg_h / q_h
    float* at        = take((size_t)2048 * cDS);
    float* qa        = take((size_t)2048 * cDS);
    float* ab        = take((size_t)6144 * cDS);
    float* preb      = take((size_t)2048 * cDS);
    float* R1        = take((size_t)32768 * 1024);        // vid_h, later pre_h
    float* R2        = take((size_t)32768 * 512);         // vid, later pre_in/ath/abh
    float* vid_h  = R1;
    float* pre_h  = R1;
    float* vid    = R2;
    float* pre_in = R2;                                   // 2048*3584 floats
    float* ath    = R2 + (size_t)2048 * cPRE;             // 2048*3072, after vid dead
    float* abh    = ath;                                  // 6144*1024, after at done

    dim3 blk(256);
    auto gemm = [&](const float* A, const float* W, const float* bias, float* C,
                    int M, int N, int K, bool relu) {
        dim3 grd((M + BM - 1) / BM, N / BN);
        if (relu) k_gemm<1><<<grd, blk, 0, stream>>>(A, W, bias, C, M, N, K);
        else      k_gemm<0><<<grd, blk, 0, stream>>>(A, W, bias, C, M, N, K);
    };

    // 1) score = softmax(paras_score)
    k_softmax16<<<1, 32, 0, stream>>>(paras_sc, score);

    // 2) vid = mlp_vit(video)   [32768,1024] -> [32768,512]
    gemm(video, Wvit1, bvit1, vid_h, 32768, 1024, 1024, true);
    gemm(vid_h, Wvit2, bvit2, vid,   32768,  512, 1024, false);

    // 3) per-segment softmax pooling -> seg_feat; then video_seg
    k_seg_pool<<<cB * cS, 512, 0, stream>>>(hand_sc, seg_start, seg_end, vid, seg_feat);
    k_wsum<<<dim3(cB, cDS / 256), blk, 0, stream>>>(score, seg_feat, video_seg);

    // 4) text_seg = score . mlp_t(para)
    gemm(para, Wt1, bt1, hbuf,   cB * cS, 3072,  cTD, true);
    gemm(hbuf, Wt2, bt2, para_f, cB * cS,  cDS, 3072, false);
    k_wsum<<<dim3(cB, cDS / 256), blk, 0, stream>>>(score, para_f, text_seg);

    // 5) vsg = score . mlp_v(video_global)
    gemm(video_glob, Wv1, bv1, hbuf, cB * cS, 3072,  cTD, true);
    gemm(hbuf, Wv2, bv2, vg,         cB * cS,  cDS, 3072, false);
    k_wsum<<<dim3(cB, cDS / 256), blk, 0, stream>>>(score, vg, vsg);

    // 6) q = mlp_t(question)
    gemm(question, Wt1, bt1, hbuf, cB, 3072,  cTD, true);
    gemm(hbuf, Wt2, bt2, qv,       cB,  cDS, 3072, false);

    // 7) at = mlp_t(a_texts); qa = q + at   (ath lives in R2 after vid is dead)
    gemm(a_texts, Wt1, bt1, ath, 2048, 3072,  cTD, true);
    gemm(ath, Wt2, bt2, at,      2048,  cDS, 3072, false);
    k_addq<<<((size_t)2048 * cDS + 255) / 256, blk, 0, stream>>>(at, qv, qa);

    // 8) ab = mlp_vit(a_buttons)
    gemm(a_buttons, Wvit1, bvit1, abh, 6144, 1024, 1024, true);
    gemm(abh, Wvit2, bvit2, ab,        6144,  cDS, 1024, false);

    // 9) pre_in concat -> mlp_pre
    {
        size_t tot = (size_t)cB * cQ * cA * cPRE;
        k_concat<<<(tot + 255) / 256, blk, 0, stream>>>(video_seg, vsg, text_seg,
                                                        qa, ab, pre_in);
    }
    gemm(pre_in, Wp1, bp1, pre_h, 2048, 14336, cPRE, true);
    gemm(pre_h, Wp2, bp2, preb,   2048,   cDS, 14336, false);

    // 10) sequential GRU + proj + argmax feedback -> logits [B,Q,A]
    k_gru<<<cB, blk, 0, stream>>>(preb, Wih, Whh, bihp, bhhp,
                                  Wpr1, bpr1, wpr2, bpr2, state0, out);
}
